// SpatioTemporalTransformer_85701777424613
// MI455X (gfx1250) — matrix operand order repaired and checked by reading, not compiled, not run on hardware
//
#include <hip/hip_runtime.h>
#include <hip/hip_bf16.h>

#define J_ 24
#define D_ 9
#define E_ 64
#define B_ 32
#define T_ 256
#define C_ (J_ * E_)   // 1536
#define OD_ (J_ * D_)  // 216

typedef __attribute__((ext_vector_type(16))) __bf16 v16bf;
typedef __attribute__((ext_vector_type(8)))  float  v8f;
typedef int i128v __attribute__((vector_size(16)));  // matches builtin's V4i

// ---------------- CDNA5 async global->LDS copy (ASYNCcnt-tracked) ----------------
#if defined(__AMDGCN__) && __has_builtin(__builtin_amdgcn_global_load_async_to_lds_b128)
#define ASYNC_LDS_COPY 1
#else
#define ASYNC_LDS_COPY 0
#endif

__device__ __forceinline__ void cp16_g2l(void* ldst, const void* gsrc) {
#if ASYNC_LDS_COPY
  __builtin_amdgcn_global_load_async_to_lds_b128(
      (__attribute__((address_space(1))) i128v*)(i128v*)(void*)gsrc,
      (__attribute__((address_space(3))) i128v*)(i128v*)ldst, 0, 0);
#else
  *(uint4*)ldst = *(const uint4*)gsrc;
#endif
}

template <int N>
__device__ __forceinline__ void wait_async() {
#if ASYNC_LDS_COPY
#if __has_builtin(__builtin_amdgcn_s_wait_asynccnt)
  __builtin_amdgcn_s_wait_asynccnt(N);
#else
  asm volatile("s_wait_asynccnt %0" ::"i"(N) : "memory");
#endif
#endif
}

// ---------------- WMMA fragment helpers (contiguous k order, same in A and B) ----------------
// A (16x32 bf16): lane m=lane&15, half lh=lane>>4; element e -> K = e + lh*16
__device__ __forceinline__ v16bf frag_a(const __bf16* base, int ld, int lane) {
  int m = lane & 15, lh = (lane >> 4) & 1;
  v16bf a;
#pragma unroll
  for (int e = 0; e < 16; ++e) {
    int k = e + lh * 16;
    a[e] = base[m * ld + k];
  }
  return a;
}
// B (32x16 bf16), source row-major K x N: element e -> K = e + lh*16, N = lane&15
__device__ __forceinline__ v16bf frag_b(const __bf16* base, int ld, int lane) {
  int n = lane & 15, lh = (lane >> 4) & 1;
  v16bf b;
#pragma unroll
  for (int e = 0; e < 16; ++e) {
    int k = e + lh * 16;
    b[e] = base[k * ld + n];
  }
  return b;
}
// B fragment from transposed source: B[k][n] = src[n*ld + k] (e.g. K^T for QK^T)
__device__ __forceinline__ v16bf frag_bt(const __bf16* base, int ld, int lane) {
  int n = lane & 15, lh = (lane >> 4) & 1;
  v16bf b;
#pragma unroll
  for (int e = 0; e < 16; ++e) {
    int k = e + lh * 16;
    b[e] = base[n * ld + k];
  }
  return b;
}
__device__ __forceinline__ v8f wmma_bf16(v16bf a, v16bf b, v8f c) {
  return __builtin_amdgcn_wmma_f32_16x16x32_bf16(false, a, false, b, (short)0, c,
                                                 false, false);
}

// ---------------- fp32 -> bf16 convert, 4-wide ----------------
__global__ void cvt_bf16x4_k(const float* __restrict__ in, __bf16* __restrict__ out,
                             int n4) {
  int i = blockIdx.x * blockDim.x + threadIdx.x;
  if (i < n4) {
    float4 v = ((const float4*)in)[i];
    __bf16 o[4] = {(__bf16)v.x, (__bf16)v.y, (__bf16)v.z, (__bf16)v.w};
    *(uint2*)(out + (size_t)i * 4) = *(uint2*)o;
  }
}

// ---------------- embedding + positional encoding ----------------
__global__ void embed_k(const float* __restrict__ src, const float* __restrict__ eW,
                        const float* __restrict__ eB, float* __restrict__ emb32,
                        __bf16* __restrict__ emb16) {
  int idx = blockIdx.x * blockDim.x + threadIdx.x;  // ((b*T+t)*J + j)*E + e
  int e  = idx & (E_ - 1);
  int j  = (idx >> 6) % J_;
  int bt = idx / (E_ * J_);
  int t  = bt % T_;
  float acc = eB[j * E_ + e];
  const float* xr = src + (size_t)bt * OD_ + j * D_;
  const float* wr = eW + (size_t)(j * D_) * E_ + e;
#pragma unroll
  for (int d = 0; d < D_; ++d) acc += xr[d] * wr[d * E_];
  // sinusoidal PE: div = exp(-(2i)*ln(10000)/E)
  float i2  = (float)(e & ~1);
  float div = __expf(i2 * (-9.210340371976184f / (float)E_));
  float ang = (float)t * div;
  acc += (e & 1) ? __cosf(ang) : __sinf(ang);
  emb32[idx] = acc;
  emb16[idx] = (__bf16)acc;
}

// ---------------- fused per-(b,j) causal attention ----------------
// grid = B*J blocks of 256 threads (8 waves). Dynamic LDS:
//   Es/Qs/Ks/Vs: 4 * 256*64 bf16 (128 KB) + P staging: 8 * 16*32 bf16 (8 KB)
__global__ void __launch_bounds__(256) attn_k(const __bf16* __restrict__ emb16,
                                              const float* __restrict__ emb32,
                                              const __bf16* __restrict__ Wq,
                                              const __bf16* __restrict__ Wk,
                                              const __bf16* __restrict__ Wv,
                                              float* __restrict__ sum1) {
  extern __shared__ __bf16 smem[];
  __bf16* Es = smem;
  __bf16* Qs = Es + T_ * E_;
  __bf16* Ks = Qs + T_ * E_;
  __bf16* Vs = Ks + T_ * E_;
  __bf16* Ps = Vs + T_ * E_;
  const int b    = blockIdx.x / J_;
  const int j    = blockIdx.x % J_;
  const int tid  = threadIdx.x;
  const int lane = tid & 31;
  const int w    = tid >> 5;

  // stage per-(b,j) embedding tile (256 x 64) via async b128 copies
  const __bf16* esrc = emb16 + ((size_t)(b * T_) * J_ + j) * E_;
  for (int i = tid * 8; i < T_ * E_; i += 256 * 8) {
    int t = i >> 6, e = i & 63;
    cp16_g2l(Es + i, esrc + (size_t)t * C_ + e);
  }
  wait_async<0>();
  __syncthreads();

  // Q/K/V = Es @ W (256x64 @ 64x64) via WMMA, bf16 results to LDS
  const __bf16* Wmats[3] = {Wq + (size_t)j * E_ * E_, Wk + (size_t)j * E_ * E_,
                            Wv + (size_t)j * E_ * E_};
  __bf16* Outs[3] = {Qs, Ks, Vs};
#pragma unroll
  for (int mat = 0; mat < 3; ++mat) {
    for (int mb = w; mb < 16; mb += 8) {
#pragma unroll
      for (int nb = 0; nb < 4; ++nb) {
        v8f acc = {};
#pragma unroll
        for (int kc = 0; kc < 2; ++kc) {
          v16bf a  = frag_a(Es + mb * 16 * E_ + kc * 32, E_, lane);
          v16bf bf = frag_b(Wmats[mat] + kc * 32 * E_ + nb * 16, E_, lane);
          acc = wmma_bf16(a, bf, acc);
        }
        int lh2 = (lane >> 4) & 1, n2 = lane & 15;
#pragma unroll
        for (int r = 0; r < 8; ++r) {
          int row = mb * 16 + r + lh2 * 8;
          Outs[mat][row * E_ + nb * 16 + n2] = (__bf16)acc[r];
        }
      }
    }
  }
  __syncthreads();

  // flash-style causal attention; wave w handles query blocks qb = w, w+8
  __bf16* Pw = Ps + w * (16 * 32);
  const int lh = (lane >> 4) & 1, ln = lane & 15;
  for (int qb = w; qb < 16; qb += 8) {
    // Q fragments are invariant over the key loop: hoist
    v16bf aq0 = frag_a(Qs + qb * 16 * E_ + 0,  E_, lane);
    v16bf aq1 = frag_a(Qs + qb * 16 * E_ + 32, E_, lane);
    v8f O0 = {}, O1 = {}, O2 = {}, O3 = {};
    float m[8], l[8];
#pragma unroll
    for (int r = 0; r < 8; ++r) { m[r] = -3.0e38f; l[r] = 0.0f; }
    const int c2max = (qb * 16 + 15) >> 5;  // 32-key chunks needed (causal)
    for (int c2 = 0; c2 <= c2max; ++c2) {
      const int kb0 = 2 * c2, kb1 = 2 * c2 + 1;
      v8f s0 = {}, s1 = {};
      s0 = wmma_bf16(aq0, frag_bt(Ks + kb0 * 16 * E_ + 0,  E_, lane), s0);
      s0 = wmma_bf16(aq1, frag_bt(Ks + kb0 * 16 * E_ + 32, E_, lane), s0);
      const bool use1 = (kb1 <= qb);
      if (use1) {
        s1 = wmma_bf16(aq0, frag_bt(Ks + kb1 * 16 * E_ + 0,  E_, lane), s1);
        s1 = wmma_bf16(aq1, frag_bt(Ks + kb1 * 16 * E_ + 32, E_, lane), s1);
      }
      float corr[8];
#pragma unroll
      for (int r = 0; r < 8; ++r) {
        const int row  = qb * 16 + r + lh * 8;
        const int col0 = kb0 * 16 + ln;
        const int col1 = kb1 * 16 + ln;
        float v0 = s0[r] * 0.125f;                      // 1/sqrt(64)
        if (col0 > row) v0 = -3.0e38f;                  // causal mask
        float v1 = use1 ? s1[r] * 0.125f : -3.0e38f;
        if (col1 > row) v1 = -3.0e38f;
        // row max across the 16-lane half (butterfly stays within half)
        float x = fmaxf(v0, v1);
        x = fmaxf(x, __shfl_xor(x, 1, 32));
        x = fmaxf(x, __shfl_xor(x, 2, 32));
        x = fmaxf(x, __shfl_xor(x, 4, 32));
        x = fmaxf(x, __shfl_xor(x, 8, 32));
        const float mnew = fmaxf(m[r], x);
        corr[r] = __expf(m[r] - mnew);
        const float p0 = __expf(v0 - mnew);
        const float p1 = __expf(v1 - mnew);
        float srow = p0 + p1;
        srow += __shfl_xor(srow, 1, 32);
        srow += __shfl_xor(srow, 2, 32);
        srow += __shfl_xor(srow, 4, 32);
        srow += __shfl_xor(srow, 8, 32);
        l[r] = l[r] * corr[r] + srow;
        m[r] = mnew;
        // stage P (transpose via per-wave LDS buffer): rows on M, cols on K
        Pw[(r + lh * 8) * 32 + ln]      = (__bf16)p0;
        Pw[(r + lh * 8) * 32 + ln + 16] = (__bf16)p1;
      }
      asm volatile("s_wait_dscnt 0" ::: "memory");  // intra-wave LDS RAW
#pragma unroll
      for (int r = 0; r < 8; ++r) {
        O0[r] *= corr[r]; O1[r] *= corr[r]; O2[r] *= corr[r]; O3[r] *= corr[r];
      }
      v16bf ap = frag_a(Pw, 32, lane);  // P as 16x32 A fragment
      O0 = wmma_bf16(ap, frag_b(Vs + c2 * 32 * E_ + 0,  E_, lane), O0);
      O1 = wmma_bf16(ap, frag_b(Vs + c2 * 32 * E_ + 16, E_, lane), O1);
      O2 = wmma_bf16(ap, frag_b(Vs + c2 * 32 * E_ + 32, E_, lane), O2);
      O3 = wmma_bf16(ap, frag_b(Vs + c2 * 32 * E_ + 48, E_, lane), O3);
    }
    // epilogue: normalize, add embedding residual, store to sum1 (B,T,J*E)
#pragma unroll
    for (int r = 0; r < 8; ++r) {
      const int trow = qb * 16 + r + lh * 8;
      const float inv = 1.0f / l[r];
      const size_t base = ((size_t)(b * T_) + trow) * C_ + j * E_;
      const float* er = emb32 + ((size_t)((b * T_ + trow) * J_) + j) * E_;
      sum1[base + 0 * 16 + ln] = O0[r] * inv + er[0 * 16 + ln];
      sum1[base + 1 * 16 + ln] = O1[r] * inv + er[1 * 16 + ln];
      sum1[base + 2 * 16 + ln] = O2[r] * inv + er[2 * 16 + ln];
      sum1[base + 3 * 16 + ln] = O3[r] * inv + er[3 * 16 + ln];
    }
  }
}

// ---------------- BatchNorm over dims (B, C) per timestep t, float4-wide ----------------
__global__ void __launch_bounds__(256) bn_k(const float* __restrict__ in,
                                            float* __restrict__ out32,
                                            __bf16* __restrict__ out16) {
  __shared__ float rs[256], rq[256];
  const int t   = blockIdx.x;
  const int tid = threadIdx.x;
  const int C4  = C_ / 4;
  float s = 0.f, q = 0.f;
  for (int i = tid; i < B_ * C4; i += 256) {
    int bi = i / C4, c = (i - bi * C4) * 4;
    float4 v = *(const float4*)&in[((size_t)bi * T_ + t) * C_ + c];
    s += v.x + v.y + v.z + v.w;
    q += v.x * v.x + v.y * v.y + v.z * v.z + v.w * v.w;
  }
  rs[tid] = s; rq[tid] = q;
  __syncthreads();
  for (int o = 128; o > 0; o >>= 1) {
    if (tid < o) { rs[tid] += rs[tid + o]; rq[tid] += rq[tid + o]; }
    __syncthreads();
  }
  const float mean = rs[0] / (float)(B_ * C_);
  const float var  = rq[0] / (float)(B_ * C_) - mean * mean;
  const float inv  = rsqrtf(var + 1e-5f);
  for (int i = tid; i < B_ * C4; i += 256) {
    int bi = i / C4, c = (i - bi * C4) * 4;
    size_t off = ((size_t)bi * T_ + t) * C_ + c;
    float4 v = *(const float4*)&in[off];
    v.x = (v.x - mean) * inv; v.y = (v.y - mean) * inv;
    v.z = (v.z - mean) * inv; v.w = (v.w - mean) * inv;
    if (out32) *(float4*)&out32[off] = v;
    __bf16 o[4] = {(__bf16)v.x, (__bf16)v.y, (__bf16)v.z, (__bf16)v.w};
    *(uint2*)&out16[off] = *(uint2*)o;
  }
}

// ---------------- FC GEMM: h2pre = h1 @ fc_W + fc_b + h1 (8192x1536x1536) ----------------
// Double-buffered async-LDS pipeline: each thread issues exactly 3 async b128s per
// stage, so s_wait_asynccnt 3 after staging buffer(next) proves buffer(cur) landed.
__global__ void __launch_bounds__(256) fc_k(const __bf16* __restrict__ A16,
                                            const float* __restrict__ A32,
                                            const __bf16* __restrict__ W16,
                                            const float* __restrict__ bias,
                                            float* __restrict__ out) {
  __shared__ __bf16 As[2][128 * 32];
  __shared__ __bf16 Bs[2][32 * 64];
  const int m0 = blockIdx.x * 128;
  const int n0 = blockIdx.y * 64;
  const int tid = threadIdx.x;
  const int lane = tid & 31, w = tid >> 5;
  v8f acc[4] = {};

  auto stage = [&](int buf, int k0) {
#pragma unroll
    for (int i = tid * 8; i < 128 * 32; i += 256 * 8) {  // 2 issues
      int r = i >> 5, c = i & 31;
      cp16_g2l(&As[buf][i], &A16[(size_t)(m0 + r) * C_ + k0 + c]);
    }
    {  // 1 issue
      int i = tid * 8;
      int r = i >> 6, c = i & 63;
      cp16_g2l(&Bs[buf][i], &W16[(size_t)(k0 + r) * C_ + n0 + c]);
    }
  };

  stage(0, 0);
  int cur = 0;
  for (int k0 = 0; k0 < C_; k0 += 32) {
    if (k0 + 32 < C_) {
      stage(cur ^ 1, k0 + 32);
      wait_async<3>();  // drain everything except the 3 just issued
    } else {
      wait_async<0>();
    }
    __syncthreads();
    v16bf a = frag_a(As[cur] + w * 16 * 32, 32, lane);
#pragma unroll
    for (int nb = 0; nb < 4; ++nb)
      acc[nb] = wmma_bf16(a, frag_b(Bs[cur] + nb * 16, 64, lane), acc[nb]);
    __syncthreads();
    cur ^= 1;
  }
  const int lh = (lane >> 4) & 1, ln = lane & 15;
#pragma unroll
  for (int nb = 0; nb < 4; ++nb) {
#pragma unroll
    for (int r = 0; r < 8; ++r) {
      int mm = m0 + w * 16 + r + lh * 8;
      int nn = n0 + nb * 16 + ln;
      size_t off = (size_t)mm * C_ + nn;
      out[off] = acc[nb][r] + bias[nn] + A32[off];
    }
  }
}

// ---------------- output GEMM: d_out = h2 @ out_W + out_b + src (8192x216x1536) ----------------
__global__ void __launch_bounds__(256) out_k(const __bf16* __restrict__ A16,
                                             const __bf16* __restrict__ W16,
                                             const float* __restrict__ bias,
                                             const float* __restrict__ src,
                                             float* __restrict__ out) {
  __shared__ __bf16 As[128 * 32];
  __shared__ __bf16 Bs[32 * 64];
  const int m0 = blockIdx.x * 128;
  const int n0 = blockIdx.y * 64;
  const int tid = threadIdx.x;
  const int lane = tid & 31, w = tid >> 5;
  v8f acc[4] = {};
  for (int k0 = 0; k0 < C_; k0 += 32) {
#pragma unroll
    for (int i = tid * 8; i < 128 * 32; i += 256 * 8) {
      int r = i >> 5, c = i & 31;
      cp16_g2l(&As[i], &A16[(size_t)(m0 + r) * C_ + k0 + c]);
    }
    {
      int i = tid * 8;
      int r = i >> 6, c = i & 63;
      int nn = n0 + c;  // 8-chunks never straddle OD_ (216 % 8 == 0)
      if (nn < OD_) {
        cp16_g2l(&Bs[i], &W16[(size_t)(k0 + r) * OD_ + nn]);
      } else {
        uint4 z = {0, 0, 0, 0};
        *(uint4*)&Bs[i] = z;
      }
    }
    wait_async<0>();
    __syncthreads();
    v16bf a = frag_a(As + w * 16 * 32, 32, lane);
#pragma unroll
    for (int nb = 0; nb < 4; ++nb)
      acc[nb] = wmma_bf16(a, frag_b(Bs + nb * 16, 64, lane), acc[nb]);
    __syncthreads();
  }
  const int lh = (lane >> 4) & 1, ln = lane & 15;
#pragma unroll
  for (int nb = 0; nb < 4; ++nb) {
#pragma unroll
    for (int r = 0; r < 8; ++r) {
      int mm = m0 + w * 16 + r + lh * 8;
      int nn = n0 + nb * 16 + ln;
      if (nn < OD_) {
        size_t off = (size_t)mm * OD_ + nn;
        out[off] = acc[nb][r] + bias[nn] + src[off];
      }
    }
  }
}

// ---------------- launch ----------------
extern "C" void kernel_launch(void* const* d_in, const int* in_sizes, int n_in,
                              void* d_out, int out_size, void* d_ws, size_t ws_size,
                              hipStream_t stream) {
  (void)in_sizes; (void)n_in; (void)out_size; (void)ws_size;
  const float* src  = (const float*)d_in[0];
  const float* embW = (const float*)d_in[1];
  const float* embB = (const float*)d_in[2];
  const float* Wq   = (const float*)d_in[3];
  const float* Wk   = (const float*)d_in[4];
  const float* Wv   = (const float*)d_in[5];
  const float* fcW  = (const float*)d_in[6];
  const float* fcB  = (const float*)d_in[7];
  const float* outW = (const float*)d_in[8];
  const float* outB = (const float*)d_in[9];
  float* out = (float*)d_out;

  char* ws = (char*)d_ws;
  size_t off = 0;
  auto alloc = [&](size_t bytes) -> void* {
    void* p = ws + off;
    off = (off + bytes + 255) & ~(size_t)255;
    return p;
  };
  const size_t N = (size_t)B_ * T_ * C_;  // == B*T*J*E
  float*  emb32 = (float*)alloc(N * 4);
  float*  sum1  = (float*)alloc(N * 4);
  float*  h1_32 = (float*)alloc(N * 4);
  float*  h2pre = (float*)alloc(N * 4);
  __bf16* emb16 = (__bf16*)alloc(N * 2);
  __bf16* h1_16 = (__bf16*)alloc(N * 2);
  __bf16* h2_16 = (__bf16*)alloc(N * 2);
  __bf16* Wq16  = (__bf16*)alloc((size_t)J_ * E_ * E_ * 2);
  __bf16* Wk16  = (__bf16*)alloc((size_t)J_ * E_ * E_ * 2);
  __bf16* Wv16  = (__bf16*)alloc((size_t)J_ * E_ * E_ * 2);
  __bf16* fcW16 = (__bf16*)alloc((size_t)C_ * C_ * 2);
  __bf16* oW16  = (__bf16*)alloc((size_t)C_ * OD_ * 2);

  const int nW4 = (J_ * E_ * E_) / 4;
  cvt_bf16x4_k<<<(nW4 + 255) / 256, 256, 0, stream>>>(Wq, Wq16, nW4);
  cvt_bf16x4_k<<<(nW4 + 255) / 256, 256, 0, stream>>>(Wk, Wk16, nW4);
  cvt_bf16x4_k<<<(nW4 + 255) / 256, 256, 0, stream>>>(Wv, Wv16, nW4);
  cvt_bf16x4_k<<<(C_ * C_ / 4 + 255) / 256, 256, 0, stream>>>(fcW, fcW16, C_ * C_ / 4);
  cvt_bf16x4_k<<<(C_ * OD_ / 4 + 255) / 256, 256, 0, stream>>>(outW, oW16, C_ * OD_ / 4);

  embed_k<<<(int)(N / 256), 256, 0, stream>>>(src, embW, embB, emb32, emb16);

  const size_t attn_lds = (size_t)(4 * T_ * E_ + 8 * 16 * 32) * sizeof(__bf16);
  attn_k<<<B_ * J_, 256, attn_lds, stream>>>(emb16, emb32, Wq16, Wk16, Wv16, sum1);

  bn_k<<<T_, 256, 0, stream>>>(sum1, h1_32, h1_16);

  fc_k<<<dim3((B_ * T_) / 128, C_ / 64), 256, 0, stream>>>(h1_16, h1_32, fcW16, fcB, h2pre);

  bn_k<<<T_, 256, 0, stream>>>(h2pre, nullptr, h2_16);

  out_k<<<dim3((B_ * T_) / 128, (OD_ + 63) / 64), 256, 0, stream>>>(h2_16, oW16, outB, src, out);
}
